// StereoBM_65944927862921
// MI455X (gfx1250) — matrix operand order repaired and checked by reading, not compiled, not run on hardware
//
#include <hip/hip_runtime.h>
#include <stdint.h>

// ---------------------------------------------------------------------------
// StereoBM on MI455X (gfx1250, wave32)
//   K1: row-wise 15-tap SAD via sliding window -> fp16 rowSAD stored
//       COLUMN-MAJOR rs[b][d][x][y] (y contiguous) in d_ws.
//   K2: vertical 15-tap box sum as ONE v_wmma_f32_16x16x32_f16 per (tile, d):
//         cost_tile(16x16) = A(16x32 banded ones, f16) x B(32x16 rowSAD, f16)
//       B slabs staged global->LDS with global_load_async_to_lds_b128
//       (double buffered, s_wait_asynccnt); column-major layout makes each
//       lane's B operand two contiguous 16B ds_load_b128 reads.
//       Elementwise running argmin over d in the WMMA C/D register layout.
// ---------------------------------------------------------------------------

typedef __attribute__((ext_vector_type(16))) _Float16 v16h;
typedef __attribute__((ext_vector_type(8)))  _Float16 h8;
typedef __attribute__((ext_vector_type(8)))  float    v8f;
typedef int v4i __attribute__((vector_size(16)));   // builtin param type

#define SB_H 544
#define SB_W 960
#define SB_D 64
#define SB_K 15
#define SB_P 7
#define SB_PAD 64   // halfs of front padding in workspace (negative-row reads)

// -------------------- async global->LDS copy (16B per lane) ----------------
#if defined(__has_builtin)
#if __has_builtin(__builtin_amdgcn_global_load_async_to_lds_b128)
#define SB_HAS_ASYNC 1
#endif
#if __has_builtin(__builtin_amdgcn_s_wait_asynccnt)
#define SB_HAS_WAITASYNC 1
#endif
#endif

#ifdef SB_HAS_ASYNC
#define SB_ASYNC_CP16(gp, lp)                                                  \
  __builtin_amdgcn_global_load_async_to_lds_b128(                              \
      (__attribute__((address_space(1))) v4i *)(gp),                           \
      (__attribute__((address_space(3))) v4i *)(lp), 0, 0)
#ifdef SB_HAS_WAITASYNC
#define SB_WAIT_ASYNC(n) __builtin_amdgcn_s_wait_asynccnt(n)
#else
#define SB_WAIT_ASYNC(n) asm volatile("s_wait_asynccnt %0" ::"i"(n) : "memory")
#endif
#else
// Fallback: plain 16B global load + LDS store (still compiles everywhere).
#define SB_ASYNC_CP16(gp, lp) (*(h8 *)(lp) = *(const h8 *)(gp))
#define SB_WAIT_ASYNC(n) asm volatile("" ::: "memory")
#endif

// ===========================================================================
// K1: rowSAD[b][d][x][y] = sum_{dx=-7..7, 0<=x+dx<W} |L[y][x+dx]-R[y][clip(x+dx-d)]|
// One wave = 32 consecutive rows of one (b,d); each lane slides across x with
// its own 15-tap window. Column-major store: 32 lanes -> 64B contiguous.
// ===========================================================================
__global__ void __launch_bounds__(256)
sb_rowsad_kernel(const float *__restrict__ L, const float *__restrict__ R,
                 _Float16 *__restrict__ rs) {
  const int t = blockIdx.x * blockDim.x + threadIdx.x;
  const int lane = t & 31;
  const int wave = t >> 5;                    // 0 .. 2*64*17-1
  const int yg = wave % (SB_H / 32);          // 17 row-groups
  const int d  = (wave / (SB_H / 32)) % SB_D;
  const int b  = wave / ((SB_H / 32) * SB_D);
  if (b >= 2) return;

  const int y = yg * 32 + lane;
  const float *__restrict__ Lr = L + ((size_t)b * SB_H + y) * SB_W;
  const float *__restrict__ Rr = R + ((size_t)b * SB_H + y) * SB_W;
  _Float16 *__restrict__ rsCol =
      rs + (size_t)(b * SB_D + d) * SB_W * SB_H + y;   // + x*SB_H per column

  auto diffAt = [&](int c) -> float {
    if (c < 0 || c >= SB_W) return 0.0f;     // horizontal zero padding
    int rc = c - d;
    rc = rc < 0 ? 0 : rc;                    // clip shift (rc <= c <= W-1)
    return fabsf(Lr[c] - Rr[rc]);
  };

  float W = 0.0f;
#pragma unroll
  for (int c = -SB_P; c <= SB_P; ++c) W += diffAt(c);

  for (int x = 0; x < SB_W; ++x) {
    rsCol[(size_t)x * SB_H] = (_Float16)W;   // b16 store, 64B/wave contiguous
    W += diffAt(x + SB_P + 1) - diffAt(x - SB_P);
  }
}

// ===========================================================================
// K2: per 16x16 output tile (one wave32):
//   Slab rows are yy = y0-8+K, K=0..31 (keeps 16B alignment of async chunks).
//   A[M][K] = 1 iff K in [M+1, M+15] and row y0-8+K valid (else 0) -> the
//   vertical 15-tap zero-padded box sum. One wmma per disparity; running
//   argmin (strict <, ascending d == jnp.argmin first-min).
// Layouts (C/D per CDNA5 ISA; A/B use this kernel's contiguous k order):
//   A 16x32 f16 : lane<16 -> M=lane, K=0..15; lane>=16 -> M=lane-16,
//                 K=16..31; VGPR v = (K=2v low16, K=2v+1 high16).
//   B 32x16 f16 : lane<16 -> col N=lane, K=0..15; lane>=16 -> col N=lane-16,
//                 K=16..31  => contiguous in LDS slab bb[n*32 + K].
//   C/D 16x16 f32: VGPR r: lanes0-15 -> M=r; lanes16-31 -> M=r+8; N=lane%16.
// ===========================================================================
__global__ void __launch_bounds__(32)
sb_box_argmin_kernel(const _Float16 *__restrict__ rs, float *__restrict__ out) {
  const int x0 = blockIdx.x * 16;
  const int y0 = blockIdx.y * 16;
  const int b  = blockIdx.z;

  const int lane = threadIdx.x;    // 0..31
  const int hi   = lane >> 4;
  const int m    = lane & 15;      // M (for A) / N (for B,C,D)

  __shared__ __align__(16) _Float16 bb[2][16 * 32];  // [col][K], two 1KB slabs

  // ---- Banded A: A[M][K] = 1 iff K in [M+1, M+15] and row y0-8+K valid.
  v16h A;
#pragma unroll
  for (int e = 0; e < 16; ++e) {
    const int v = e >> 1, odd = e & 1;
    int K = 2 * v + odd + 16 * hi;
    const int yy = y0 - 8 + K;
    const bool on =
        (K >= m + 1) && (K <= m + SB_K) && (yy >= 0) && (yy < SB_H);
    A[e] = on ? (_Float16)1.0f : (_Float16)0.0f;
  }

  // ---- Async staging: 64 chunks of 16B = one 16-col x 32-row f16 slab.
  // chunk c: col n = c>>2, part = c&3 (8 halfs). Global is column-major so
  // each chunk is contiguous; 16B aligned because (1088B stride, y0-8+8*part).
  auto issue = [&](int d, int buf) {
    const size_t dBase = (size_t)(b * SB_D + d) * SB_W * SB_H;
#pragma unroll
    for (int i = 0; i < 2; ++i) {
      const int c = lane + 32 * i;
      const int n = c >> 2;
      const int part = c & 3;
      const _Float16 *g =
          rs + dBase + (size_t)(x0 + n) * SB_H + (y0 - 8 + part * 8);
      SB_ASYNC_CP16(g, &bb[buf][n * 32 + part * 8]);
    }
  };

  issue(0, 0);

  v8f best;
  int bestd[8];
#pragma unroll
  for (int r = 0; r < 8; ++r) { best[r] = 3.4e38f; bestd[r] = 0; }

  const int kbase = hi * 16;
  for (int d = 0; d < SB_D; ++d) {
    const int buf = d & 1;
    if (d + 1 < SB_D) {
      issue(d + 1, (d + 1) & 1);
      SB_WAIT_ASYNC(2);          // in-order completion: slab d is done
    } else {
      SB_WAIT_ASYNC(0);
    }
    asm volatile("" ::: "memory");

    // B operand: two contiguous 16B LDS reads per lane.
    const h8 blo = *(const h8 *)(&bb[buf][m * 32 + kbase]);
    const h8 bhi = *(const h8 *)(&bb[buf][m * 32 + kbase + 8]);
    const v16h B = __builtin_shufflevector(blo, bhi, 0, 1, 2, 3, 4, 5, 6, 7, 8,
                                           9, 10, 11, 12, 13, 14, 15);

    v8f c = {};
    c = __builtin_amdgcn_wmma_f32_16x16x32_f16(
        /*neg_a=*/false, A, /*neg_b=*/false, B,
        /*c_mod=*/(short)0, c, /*reuse_a=*/false, /*reuse_b=*/false);

#pragma unroll
    for (int r = 0; r < 8; ++r) {
      if (c[r] < best[r]) { best[r] = c[r]; bestd[r] = d; }  // first-min
    }
  }

  // ---- Store argmin (float, per harness output convention).
#pragma unroll
  for (int r = 0; r < 8; ++r) {
    const int y = y0 + r + 8 * hi;
    out[((size_t)b * SB_H + y) * SB_W + x0 + m] = (float)bestd[r];
  }
}

// ===========================================================================
extern "C" void kernel_launch(void *const *d_in, const int *in_sizes, int n_in,
                              void *d_out, int out_size, void *d_ws,
                              size_t ws_size, hipStream_t stream) {
  (void)in_sizes; (void)n_in; (void)out_size; (void)ws_size;
  const float *L = (const float *)d_in[0];
  const float *R = (const float *)d_in[1];
  // d_in[2]=block_size(15), d_in[3]=max_disparity(64): fixed per reference setup.
  float *out = (float *)d_out;
  // Workspace: front pad (negative-row slab reads) + 2*64*960*544 fp16 (~128MB)
  // + slack at the end for bottom-tile over-reads (finite garbage, A-masked).
  _Float16 *rs = (_Float16 *)d_ws + SB_PAD;

  const int k1_threads = 2 * SB_D * (SB_H / 32) * 32;   // 69632
  sb_rowsad_kernel<<<dim3((k1_threads + 255) / 256), dim3(256), 0, stream>>>(
      L, R, rs);

  sb_box_argmin_kernel<<<dim3(SB_W / 16, SB_H / 16, 2), dim3(32), 0, stream>>>(
      rs, out);
}
